// Attn_25451976196192
// MI455X (gfx1250) — matrix operand order repaired and checked by reading, not compiled, not run on hardware
//
#include <hip/hip_runtime.h>
#include <hip/hip_bf16.h>
#include <stdint.h>

// Problem constants (from reference): B=64, S_STATE=512, S_SEQ=2048, H=512
#define NB     64
#define NSTATE 512
#define NSEQ   2048
#define NH     512

typedef __attribute__((ext_vector_type(16))) __bf16 v16bf;
typedef __attribute__((ext_vector_type(8)))  __bf16 v8bf;
typedef __attribute__((ext_vector_type(2)))  __bf16 v2bf;
typedef __attribute__((ext_vector_type(8)))  float  v8f;

// Hardware f32 -> bf16 convert (RNE). gfx1250 legalizes __bf16 natively.
__device__ __forceinline__ unsigned short f2bfu(float f) {
    __bf16 h = (__bf16)f;
    unsigned short u; __builtin_memcpy(&u, &h, 2);
    return u;
}
// Packed pair convert: lets the backend pick v_cvt_pk_bf16_f32.
__device__ __forceinline__ unsigned int f2x2(float x, float y) {
    v2bf v; v[0] = (__bf16)x; v[1] = (__bf16)y;
    unsigned int u; __builtin_memcpy(&u, &v, 4);
    return u;
}

__device__ __forceinline__ v16bf cat16(v8bf lo, v8bf hi) {
    return __builtin_shufflevector(lo, hi, 0,1,2,3,4,5,6,7,8,9,10,11,12,13,14,15);
}

// ---------------------------------------------------------------------------
// Kernel 1: Q[m,n] = sum_k out_state[m,k] * W[k,n], output bf16.
// M = NB*NSTATE = 32768, N = K = 512. Tile 128x128, K-step 32. 8 waves.
// ---------------------------------------------------------------------------
#define K1_TM 128
#define K1_TN 128
#define KT    32
#define LDT   40   // padded LDS row stride (elements): 80 bytes, 16B-aligned runs

__global__ __launch_bounds__(256) void qgemm_kernel(
    const float* __restrict__ A,     // [32768, 512] out_state
    const float* __restrict__ W,     // [512, 512]   attn_w  (W[h,d], k=h, n=d)
    unsigned short* __restrict__ Qb) // [32768, 512] bf16 out
{
    __shared__ unsigned short sA[K1_TM * LDT];  // [m][k] bf16
    __shared__ unsigned short sB[K1_TN * LDT];  // [n][k] bf16 (W transposed)

    const int t    = threadIdx.x;
    const int wave = t >> 5;
    const int lane = t & 31;
    const int wm   = wave >> 2;          // 0..1  -> 64-row half
    const int wn   = wave & 3;           // 0..3  -> 32-col slice
    const int r16  = lane & 15;
    const int kh   = lane >> 4;          // K-half select

    const int m0 = (blockIdx.x >> 2) * K1_TM;     // 256 M-tiles
    const int n0 = (blockIdx.x & 3) * K1_TN;      // 4   N-tiles

    v8f acc[4][2] = {};

    for (int k0 = 0; k0 < NH; k0 += KT) {
        __syncthreads();
        // Stage A tile 128x32 (f32 -> bf16). 2 threads/row, 16 floats each,
        // packed-pair converts, 8-byte LDS stores.
        {
            const int r  = t >> 1;
            const int c0 = (t & 1) * 16;
            const float* src = A + (size_t)(m0 + r) * NH + k0 + c0;
            unsigned short* dst = &sA[r * LDT + c0];
#pragma unroll
            for (int i = 0; i < 16; i += 4) {
                float4 v = *reinterpret_cast<const float4*>(src + i);
                uint2 p;
                p.x = f2x2(v.x, v.y);
                p.y = f2x2(v.z, v.w);
                *reinterpret_cast<uint2*>(dst + i) = p;
            }
        }
        // Stage W tile 32x128 transposed into sB[n][k] (coalesced global reads).
        {
#pragma unroll
            for (int i = 0; i < 16; ++i) {
                int e  = i * 256 + t;
                int kk = e >> 7;           // 0..31
                int nn = e & 127;          // 0..127
                sB[nn * LDT + kk] = f2bfu(W[(size_t)(k0 + kk) * NH + n0 + nn]);
            }
        }
        __syncthreads();

        // A fragments: lane r16 holds row m, K runs {kh*8..+7, 16+kh*8..+7}
        v16bf af[4];
#pragma unroll
        for (int mt = 0; mt < 4; ++mt) {
            const int base = (wm * 64 + mt * 16 + r16) * LDT;
            v8bf lo = *reinterpret_cast<const v8bf*>(&sA[base + kh * 8]);
            v8bf hi = *reinterpret_cast<const v8bf*>(&sA[base + 16 + kh * 8]);
            af[mt] = cat16(lo, hi);
        }
        // B fragments: lane r16 = column n, K runs {kh*8..+7, 16+kh*8..+7}
        v16bf bfr[2];
#pragma unroll
        for (int nt = 0; nt < 2; ++nt) {
            const int base = (wn * 32 + nt * 16 + r16) * LDT + kh * 8;
            v8bf lo = *reinterpret_cast<const v8bf*>(&sB[base]);
            v8bf hi = *reinterpret_cast<const v8bf*>(&sB[base + 16]);
            bfr[nt] = cat16(lo, hi);
        }
#pragma unroll
        for (int mt = 0; mt < 4; ++mt)
#pragma unroll
            for (int nt = 0; nt < 2; ++nt)
                acc[mt][nt] = __builtin_amdgcn_wmma_f32_16x16x32_bf16(
                    false, af[mt], false, bfr[nt], (short)0, acc[mt][nt], false, false);
    }

    // Store Q as bf16. C layout: M = mt*16 + kh*8 + r, N = nt*16 + r16.
#pragma unroll
    for (int mt = 0; mt < 4; ++mt)
#pragma unroll
        for (int nt = 0; nt < 2; ++nt)
#pragma unroll
            for (int r = 0; r < 8; ++r) {
                int m = m0 + wm * 64 + mt * 16 + kh * 8 + r;
                int n = n0 + wn * 32 + nt * 16 + r16;
                Qb[(size_t)m * NH + n] = f2bfu(acc[mt][nt][r]);
            }
}

// ---------------------------------------------------------------------------
// Kernel 2: S[b, i, j] = Q[b,i,:] . history[b,j,:]  (raw energies -> d_out)
// plus online per-row softmax stats (m, s) -> workspace.
// Block handles 64 rows of one batch; loops j in 256-col tiles. 8 waves,
// each wave = 64 rows x 32 cols = 8 WMMA tiles.
// ---------------------------------------------------------------------------
#define K2_TM 64
#define K2_TN 256

__global__ __launch_bounds__(256) void attn_energy_kernel(
    const unsigned short* __restrict__ Qb,   // [B*S_STATE, H] bf16
    const float* __restrict__ hist,          // [B, S_SEQ, H] f32
    float* __restrict__ outE,                // [B, S_STATE, S_SEQ] f32 (raw)
    float* __restrict__ statM,               // [B*S_STATE]
    float* __restrict__ statS)               // [B*S_STATE]
{
    __shared__ unsigned short sQ[K2_TM * LDT];   // [i][k]
    __shared__ unsigned short sH[K2_TN * LDT];   // [j][k]
    __shared__ float pMax[K2_TM][8];
    __shared__ float pSum[K2_TM][8];
    __shared__ float mNew[K2_TM];
    __shared__ float mRun[K2_TM];
    __shared__ float sRun[K2_TM];

    const int t    = threadIdx.x;
    const int wave = t >> 5;                 // 0..7 -> 32-col slice
    const int lane = t & 31;
    const int r16  = lane & 15;
    const int kh   = lane >> 4;

    const int b  = blockIdx.x >> 3;
    const int i0 = (blockIdx.x & 7) * K2_TM;
    const size_t qbase = ((size_t)b * NSTATE + i0) * NH;   // bf16 elems
    const size_t hbase = (size_t)b * NSEQ * NH;            // f32 elems
    const size_t obase = ((size_t)b * NSTATE + i0) * NSEQ; // f32 elems

    if (t < K2_TM) { mRun[t] = -3.0e38f; sRun[t] = 0.0f; }
    __syncthreads();

    for (int j0 = 0; j0 < NSEQ; j0 += K2_TN) {
        v8f acc[4][2] = {};

        for (int k0 = 0; k0 < NH; k0 += KT) {
            __syncthreads();
            // Stage Q tile 64x32 bf16 (raw copy, 16B per thread).
            {
                const int r = t >> 2;
                const int c = (t & 3) * 8;
                uint4 v = *reinterpret_cast<const uint4*>(Qb + qbase + (size_t)r * NH + k0 + c);
                *reinterpret_cast<uint4*>(&sQ[r * LDT + c]) = v;
            }
            // Stage history tile 256x32: float2 loads, packed bf16 pair,
            // one b32 LDS store per pair. 16 pairs/thread, coalesced.
#pragma unroll 4
            for (int i = 0; i < 16; ++i) {
                int p = i * 256 + t;           // pair index in [0, 4096)
                int r = p >> 4;                // 16 pairs per 32-elem row
                int c = (p & 15) * 2;
                float2 v = *reinterpret_cast<const float2*>(
                    hist + hbase + (size_t)(j0 + r) * NH + k0 + c);
                *reinterpret_cast<unsigned int*>(&sH[r * LDT + c]) = f2x2(v.x, v.y);
            }
            // Prefetch next K-slab of history (gfx1250 global_prefetch_b8).
            if (k0 + KT < NH)
                __builtin_prefetch(&hist[hbase + (size_t)(j0 + t) * NH + k0 + KT], 0, 1);
            __syncthreads();

            v16bf qf[4];
#pragma unroll
            for (int mt = 0; mt < 4; ++mt) {
                const int base = (mt * 16 + r16) * LDT;
                v8bf lo = *reinterpret_cast<const v8bf*>(&sQ[base + kh * 8]);
                v8bf hi = *reinterpret_cast<const v8bf*>(&sQ[base + 16 + kh * 8]);
                qf[mt] = cat16(lo, hi);
            }
            v16bf hf[2];
#pragma unroll
            for (int nt = 0; nt < 2; ++nt) {
                const int base = (wave * 32 + nt * 16 + r16) * LDT + kh * 8;
                v8bf lo = *reinterpret_cast<const v8bf*>(&sH[base]);
                v8bf hi = *reinterpret_cast<const v8bf*>(&sH[base + 16]);
                hf[nt] = cat16(lo, hi);
            }
#pragma unroll
            for (int mt = 0; mt < 4; ++mt)
#pragma unroll
                for (int nt = 0; nt < 2; ++nt)
                    acc[mt][nt] = __builtin_amdgcn_wmma_f32_16x16x32_bf16(
                        false, qf[mt], false, hf[nt], (short)0, acc[mt][nt], false, false);
        }

        // ---- online softmax stats over this 64x256 tile ----
        // 1) per-wave row-max partials (reduce across 16 lanes of each half)
#pragma unroll
        for (int mt = 0; mt < 4; ++mt)
#pragma unroll
            for (int r = 0; r < 8; ++r) {
                float v = fmaxf(acc[mt][0][r], acc[mt][1][r]);
                for (int off = 1; off < 16; off <<= 1)
                    v = fmaxf(v, __shfl_xor(v, off, 32));
                if (r16 == 0) pMax[mt * 16 + kh * 8 + r][wave] = v;
            }
        __syncthreads();
        if (t < K2_TM) {
            float v = pMax[t][0];
#pragma unroll
            for (int w = 1; w < 8; ++w) v = fmaxf(v, pMax[t][w]);
            mNew[t] = fmaxf(mRun[t], v);
        }
        __syncthreads();
        // 2) per-wave exp-sum partials against mNew
#pragma unroll
        for (int mt = 0; mt < 4; ++mt)
#pragma unroll
            for (int r = 0; r < 8; ++r) {
                const int mrow = mt * 16 + kh * 8 + r;
                const float mn = mNew[mrow];
                float p = __expf(acc[mt][0][r] - mn) + __expf(acc[mt][1][r] - mn);
                for (int off = 1; off < 16; off <<= 1)
                    p += __shfl_xor(p, off, 32);
                if (r16 == 0) pSum[mrow][wave] = p;
            }
        __syncthreads();
        if (t < K2_TM) {
            float s = 0.0f;
#pragma unroll
            for (int w = 0; w < 8; ++w) s += pSum[t][w];
            sRun[t] = sRun[t] * __expf(mRun[t] - mNew[t]) + s;
            mRun[t] = mNew[t];
        }
        // 3) store raw energies
#pragma unroll
        for (int mt = 0; mt < 4; ++mt)
#pragma unroll
            for (int nt = 0; nt < 2; ++nt)
#pragma unroll
                for (int r = 0; r < 8; ++r) {
                    int m = mt * 16 + kh * 8 + r;
                    int n = wave * 32 + nt * 16 + r16;
                    outE[obase + (size_t)m * NSEQ + j0 + n] = acc[mt][nt][r];
                }
        // next j-iteration's first __syncthreads() fences pMax/pSum reuse
    }

    __syncthreads();
    if (t < K2_TM) {
        statM[(size_t)b * NSTATE + i0 + t] = mRun[t];
        statS[(size_t)b * NSTATE + i0 + t] = sRun[t];
    }
}

// ---------------------------------------------------------------------------
// Kernel 3: normalize in place: out = exp(x - m) / s   (float4 per thread)
// ---------------------------------------------------------------------------
__global__ __launch_bounds__(256) void softmax_norm_kernel(
    float* __restrict__ outE,
    const float* __restrict__ statM,
    const float* __restrict__ statS)
{
    const size_t idx  = (size_t)blockIdx.x * 256 + threadIdx.x; // one float4 each
    const size_t base = idx * 4;
    const size_t row  = base >> 11;                             // / NSEQ (2048)
    const float m   = statM[row];
    const float inv = 1.0f / statS[row];
    float4 v = *reinterpret_cast<float4*>(outE + base);
    v.x = __expf(v.x - m) * inv;
    v.y = __expf(v.y - m) * inv;
    v.z = __expf(v.z - m) * inv;
    v.w = __expf(v.w - m) * inv;
    *reinterpret_cast<float4*>(outE + base) = v;
}

// ---------------------------------------------------------------------------
extern "C" void kernel_launch(void* const* d_in, const int* in_sizes, int n_in,
                              void* d_out, int out_size, void* d_ws, size_t ws_size,
                              hipStream_t stream) {
    const float* out_state = (const float*)d_in[0];  // [64, 512, 512]
    const float* history   = (const float*)d_in[1];  // [64, 2048, 512]
    const float* attn_w    = (const float*)d_in[2];  // [512, 512]
    (void)in_sizes; (void)n_in; (void)out_size; (void)ws_size;
    // attn_b (d_in[3]) is provably irrelevant: it shifts every energy in a
    // softmax row by the same constant, and softmax is shift-invariant.

    // Workspace layout: Q bf16 [32768 x 512] = 33,554,432 B, then stats 2x128KB.
    unsigned short* Qb = (unsigned short*)d_ws;
    float* statM = (float*)((char*)d_ws + (size_t)NB * NSTATE * NH * 2);
    float* statS = statM + (size_t)NB * NSTATE;
    float* outE  = (float*)d_out;

    // 1) Q = out_state @ W  (bf16 output): 256 M-tiles x 4 N-tiles
    qgemm_kernel<<<1024, 256, 0, stream>>>(out_state, attn_w, Qb);

    // 2) raw energies + online softmax stats: 64 batches x 8 row-tiles
    attn_energy_kernel<<<512, 256, 0, stream>>>(Qb, history, outE, statM, statS);

    // 3) normalize: 64*512*2048 / (4*256) blocks
    softmax_norm_kernel<<<65536, 256, 0, stream>>>(outE, statM, statS);
}
